// MultiHeadAttention_12575664242956
// MI455X (gfx1250) — matrix operand order repaired and checked by reading, not compiled, not run on hardware
//
#include <hip/hip_runtime.h>

typedef __attribute__((ext_vector_type(16))) __bf16 v16bf;
typedef __attribute__((ext_vector_type(8)))  __bf16 v8bf;
typedef __attribute__((ext_vector_type(8)))  float  v8f;
typedef __attribute__((ext_vector_type(4)))  unsigned int u32x4;
typedef __attribute__((ext_vector_type(8)))  int i32x8;
typedef __attribute__((ext_vector_type(4)))  int i32x4;

#define B_  8
#define T_  2048
#define E_  384
#define H_  6
#define HS_ 64
#define NQK 384   // H_*HS_

__device__ __forceinline__ v8f wmma_bf16(v16bf a, v16bf b, v8f c) {
    return __builtin_amdgcn_wmma_f32_16x16x32_bf16(false, a, false, b, (short)0, c, false, false);
}

__device__ __forceinline__ v8f zero8() {
    v8f z = {0.f,0.f,0.f,0.f,0.f,0.f,0.f,0.f};
    return z;
}

// A-fragment (16x32 bf16): lane L (m=L&15, sel=L>>4) holds row m,
// K = {sel*16..sel*16+15}  (two contiguous 16B chunks)
__device__ __forceinline__ v16bf load_a_frag(const __bf16* row, int k0, int sel) {
    union { v16bf v; v8bf h[2]; } u;
    u.h[0] = *(const v8bf*)(row + k0 + sel * 16);
    u.h[1] = *(const v8bf*)(row + k0 + sel * 16 + 8);
    return u.v;
}

__device__ __forceinline__ float redmax16(float v) {
    #pragma unroll
    for (int off = 1; off < 16; off <<= 1) v = fmaxf(v, __shfl_xor(v, off, 16));
    return v;
}
__device__ __forceinline__ float redsum16(float v) {
    #pragma unroll
    for (int off = 1; off < 16; off <<= 1) v += __shfl_xor(v, off, 16);
    return v;
}

// ---------------- Tensor Data Mover: 2D tile global -> LDS ----------------
// D# per CDNA5 ISA: group0 {flags, lds_addr, gaddr_lo,
// gaddr_hi|type=2}, group1 {data_size=2B, tensor/tile dims, dim0 stride}.
// tensor dims == tile dims (tile fully in-bounds). Groups 2/3 zero (2-D).
__device__ __forceinline__ void tdm_load_2d(const __bf16* lds_dst, const __bf16* gsrc,
                                            unsigned tile_d0, unsigned tile_d1,
                                            unsigned stride0_elems) {
    unsigned long long ga = (unsigned long long)(size_t)gsrc;
    unsigned lds_off = (unsigned)(size_t)lds_dst;          // low 32 bits = LDS byte offset
    u32x4 g0;
    g0[0] = 1u;                                            // count=1 (valid user D#)
    g0[1] = lds_off;
    g0[2] = (unsigned)(ga & 0xFFFFFFFFu);
    g0[3] = (unsigned)((ga >> 32) & 0x01FFFFFFu) | (2u << 30);   // addr[56:32] | type=2
    i32x8 g1;
    g1[0] = 1 << 16;                                       // wg_mask=0, data_size=1 (2 bytes)
    g1[1] = (int)((tile_d0 & 0xFFFFu) << 16);              // tensor_dim0[15:0]
    g1[2] = (int)((tile_d1 & 0xFFFFu) << 16);              // tensor_dim0 hi=0 | tensor_dim1[15:0]
    g1[3] = (int)((tile_d0 & 0xFFFFu) << 16);              // tensor_dim1 hi=0 | tile_dim0
    g1[4] = (int)(tile_d1 & 0xFFFFu);                      // tile_dim1 | tile_dim2=0
    g1[5] = (int)stride0_elems;                            // tensor_dim0_stride[31:0]
    g1[6] = 0;
    g1[7] = 0;
    i32x4 z4 = {0, 0, 0, 0};
#if __has_include(<hip/amd_detail/amd_gfx1250_TDM.h>)
    i32x8 z8 = {0, 0, 0, 0, 0, 0, 0, 0};
    __builtin_amdgcn_tensor_load_to_lds(g0, g1, z4, z4, z8, 0);   // clang-23 lane (6-arg)
#else
    __builtin_amdgcn_tensor_load_to_lds(g0, g1, z4, z4, 0);       // ROCm 7.2 lane (5-arg)
#endif
}

// ---------------- prep kernels ----------------
__global__ void k_cvt_x(const float* __restrict__ x, __bf16* __restrict__ xb, int n) {
    int i = blockIdx.x * blockDim.x + threadIdx.x;
    if (i < n) xb[i] = (__bf16)x[i];
}

// WqT/WkT/WvT: [n = h*64+d][e] ;  WoT: [e][k = h*64+d]
__global__ void k_prep_w(const float* __restrict__ Wq, const float* __restrict__ Wk,
                         const float* __restrict__ Wv, const float* __restrict__ Wo,
                         __bf16* __restrict__ WqT, __bf16* __restrict__ WkT,
                         __bf16* __restrict__ WvT, __bf16* __restrict__ WoT) {
    int idx = blockIdx.x * blockDim.x + threadIdx.x;
    if (idx >= 384 * 384) return;
    int i = idx / 384, j = idx % 384;
    int src = ((i >> 6) * 384 + j) * 64 + (i & 63);   // Wq[h][e][d], h=i/64, d=i%64, e=j
    WqT[idx] = (__bf16)Wq[src];
    WkT[idx] = (__bf16)Wk[src];
    WvT[idx] = (__bf16)Wv[src];
    WoT[idx] = (__bf16)Wo[j * 384 + i];               // Wo flat [k][e] -> WoT[e][k]
}

// ---------------- fused QKV projection GEMM ----------------
// grid (N/64=6, M/64=256), block 128 (4 waves); wave tile = 16M x 64N
__global__ __launch_bounds__(128)
void k_qkv(const __bf16* __restrict__ xb,
           const __bf16* __restrict__ WqT, const __bf16* __restrict__ WkT,
           const __bf16* __restrict__ WvT,
           __bf16* __restrict__ q, __bf16* __restrict__ k, __bf16* __restrict__ vT) {
    int wave = threadIdx.x >> 5, lane = threadIdx.x & 31;
    int m_loc = lane & 15, sel = lane >> 4;
    int mbase = (blockIdx.y * 4 + wave) * 16;
    int nbase = blockIdx.x * 64;

    v8f aq[4], ak[4], av[4];
    #pragma unroll
    for (int nt = 0; nt < 4; ++nt) { aq[nt] = zero8(); ak[nt] = zero8(); av[nt] = zero8(); }

    const __bf16* arow = xb + (size_t)(mbase + m_loc) * E_;
    for (int k0 = 0; k0 < E_; k0 += 32) {
        v16bf a = load_a_frag(arow, k0, sel);
        #pragma unroll
        for (int nt = 0; nt < 4; ++nt) {
            size_t boff = (size_t)(nbase + nt * 16 + m_loc) * E_ + k0 + sel * 16;
            v16bf bq = *(const v16bf*)(WqT + boff);
            v16bf bk = *(const v16bf*)(WkT + boff);
            v16bf bv = *(const v16bf*)(WvT + boff);
            aq[nt] = wmma_bf16(a, bq, aq[nt]);
            ak[nt] = wmma_bf16(a, bk, ak[nt]);
            av[nt] = wmma_bf16(a, bv, av[nt]);
        }
    }

    int h = blockIdx.x;                    // nbase/64
    #pragma unroll
    for (int r = 0; r < 8; ++r) {
        int row = mbase + r + 8 * sel;
        int b = row >> 11, t = row & (T_ - 1);
        size_t bh = (size_t)(b * H_ + h);
        #pragma unroll
        for (int nt = 0; nt < 4; ++nt) {
            int d = nt * 16 + m_loc;
            q [(bh * T_ + t) * HS_ + d] = (__bf16)aq[nt][r];
            k [(bh * T_ + t) * HS_ + d] = (__bf16)ak[nt][r];
            vT[(bh * HS_ + d) * T_ + t] = (__bf16)av[nt][r];
        }
    }
}

// ---------------- flash-style attention, TDM double-buffered K/V ----------------
// grid (B*H=48, T/64=32), block 128 (4 waves); wave = one 16-row query tile.
// Wave 0 DMAs the shared K tile [32][64] and V^T tile [64][32] for the whole
// block via tensor_load_to_lds; compute overlaps the next tile's DMA.
__global__ __launch_bounds__(128)
void k_attn(const __bf16* __restrict__ q, const __bf16* __restrict__ k,
            const __bf16* __restrict__ vT, __bf16* __restrict__ ctx) {
    __shared__ __align__(32) __bf16 ldsK[2][32 * 64];
    __shared__ __align__(32) __bf16 ldsV[2][64 * 32];
    __shared__ __align__(32) __bf16 pbuf[4][16 * 32];

    int bh = blockIdx.x;
    int wave = threadIdx.x >> 5, lane = threadIdx.x & 31;
    int m_loc = lane & 15, sel = lane >> 4;
    int ibase = blockIdx.y * 64 + wave * 16;

    const __bf16* Qh = q  + (size_t)bh * T_ * HS_;
    const __bf16* Kh = k  + (size_t)bh * T_ * HS_;
    const __bf16* Vh = vT + (size_t)bh * HS_ * T_;

    // Q fragments resident (16 x 64 = 2 K-steps of 32)
    v16bf qa[2];
    #pragma unroll
    for (int kk = 0; kk < 2; ++kk)
        qa[kk] = load_a_frag(Qh + (size_t)(ibase + m_loc) * HS_, kk * 32, sel);

    v8f o[4];
    #pragma unroll
    for (int f = 0; f < 4; ++f) o[f] = zero8();
    float mrow[8], lrow[8];
    #pragma unroll
    for (int r = 0; r < 8; ++r) { mrow[r] = -1e30f; lrow[r] = 0.f; }

    // prologue: DMA first K/V tiles into buffer 0
    if (wave == 0) {
        tdm_load_2d(&ldsK[0][0], Kh, HS_, 32, HS_);       // 32 key rows x 64
        tdm_load_2d(&ldsV[0][0], Vh, 32, HS_, T_);        // 64 d-rows x 32 keys
    }
    int cur = 0;

    for (int j0 = 0; j0 < T_; j0 += 32) {
        __syncthreads();   // all waves done reading buf[cur^1] -> safe to overwrite
        if (wave == 0) {
            if (j0 + 32 < T_) {
                tdm_load_2d(&ldsK[cur ^ 1][0], Kh + (size_t)(j0 + 32) * HS_, HS_, 32, HS_);
                tdm_load_2d(&ldsV[cur ^ 1][0], Vh + (j0 + 32), 32, HS_, T_);
                __builtin_amdgcn_s_wait_tensorcnt(2);     // current tile's 2 DMAs done
            } else {
                __builtin_amdgcn_s_wait_tensorcnt(0);
            }
        }
        __syncthreads();   // buf[cur] published to all waves

        // S = Q K^T : two 16-key tiles, K-dim = HS = 2 x 32
        v8f s0 = zero8(), s1 = zero8();
        #pragma unroll
        for (int kk = 0; kk < 2; ++kk) {
            v16bf kb0 = *(const v16bf*)(&ldsK[cur][(     m_loc) * HS_ + kk * 32 + sel * 16]);
            v16bf kb1 = *(const v16bf*)(&ldsK[cur][(16 + m_loc) * HS_ + kk * 32 + sel * 16]);
            s0 = wmma_bf16(qa[kk], kb0, s0);
            s1 = wmma_bf16(qa[kk], kb1, s1);
        }
        // multiplicative tril mask, scale 1/8, online softmax (all T cols participate)
        #pragma unroll
        for (int r = 0; r < 8; ++r) {
            int ig = ibase + r + 8 * sel;
            float v0 = (j0 +      m_loc <= ig) ? s0[r] * 0.125f : 0.f;
            float v1 = (j0 + 16 + m_loc <= ig) ? s1[r] * 0.125f : 0.f;
            float mn = fmaxf(mrow[r], redmax16(fmaxf(v0, v1)));
            float cf = __expf(mrow[r] - mn);
            float p0 = __expf(v0 - mn);
            float p1 = __expf(v1 - mn);
            lrow[r] = lrow[r] * cf + redsum16(p0 + p1);
            mrow[r] = mn;
            #pragma unroll
            for (int f = 0; f < 4; ++f) o[f][r] *= cf;
            pbuf[wave][(r + 8 * sel) * 32 +      m_loc] = (__bf16)p0;
            pbuf[wave][(r + 8 * sel) * 32 + 16 + m_loc] = (__bf16)p1;
        }
        // D-layout -> A-layout via wave-private LDS (per-wave DS ops are in-order,
        // no block barrier needed), then O += P V  (K-dim = 32 keys)
        v16bf pa;
        {
            union { v16bf v; v8bf h[2]; } u;
            u.h[0] = *(const v8bf*)(&pbuf[wave][m_loc * 32 + sel * 16]);
            u.h[1] = *(const v8bf*)(&pbuf[wave][m_loc * 32 + sel * 16 + 8]);
            pa = u.v;
        }
        #pragma unroll
        for (int f = 0; f < 4; ++f) {
            v16bf vb = *(const v16bf*)(&ldsV[cur][(f * 16 + m_loc) * 32 + sel * 16]);
            o[f] = wmma_bf16(pa, vb, o[f]);
        }
        cur ^= 1;
    }

    int b = bh / H_, h = bh % H_;
    #pragma unroll
    for (int r = 0; r < 8; ++r) {
        float inv = 1.0f / lrow[r];
        int t = ibase + r + 8 * sel;
        size_t rowoff = ((size_t)b * T_ + t) * NQK + h * HS_;
        #pragma unroll
        for (int f = 0; f < 4; ++f)
            ctx[rowoff + f * 16 + m_loc] = (__bf16)(o[f][r] * inv);
    }
}

// ---------------- output projection GEMM (head-sum = K dim) ----------------
// grid (N/64=6, M/64=256), block 128; out f32 written directly
__global__ __launch_bounds__(128)
void k_out(const __bf16* __restrict__ ctx, const __bf16* __restrict__ WoT,
           float* __restrict__ out) {
    int wave = threadIdx.x >> 5, lane = threadIdx.x & 31;
    int m_loc = lane & 15, sel = lane >> 4;
    int mbase = (blockIdx.y * 4 + wave) * 16;
    int nbase = blockIdx.x * 64;

    v8f acc[4];
    #pragma unroll
    for (int nt = 0; nt < 4; ++nt) acc[nt] = zero8();

    const __bf16* arow = ctx + (size_t)(mbase + m_loc) * NQK;
    for (int k0 = 0; k0 < NQK; k0 += 32) {
        v16bf a = load_a_frag(arow, k0, sel);
        #pragma unroll
        for (int nt = 0; nt < 4; ++nt) {
            v16bf bw = *(const v16bf*)(WoT + (size_t)(nbase + nt * 16 + m_loc) * NQK + k0 + sel * 16);
            acc[nt] = wmma_bf16(a, bw, acc[nt]);
        }
    }
    #pragma unroll
    for (int r = 0; r < 8; ++r) {
        size_t row = (size_t)(mbase + r + 8 * sel);
        #pragma unroll
        for (int nt = 0; nt < 4; ++nt)
            out[row * E_ + nbase + nt * 16 + m_loc] = acc[nt][r];
    }
}

extern "C" void kernel_launch(void* const* d_in, const int* in_sizes, int n_in,
                              void* d_out, int out_size, void* d_ws, size_t ws_size,
                              hipStream_t stream) {
    const float* x  = (const float*)d_in[0];
    const float* Wq = (const float*)d_in[1];
    const float* Wk = (const float*)d_in[2];
    const float* Wv = (const float*)d_in[3];
    const float* Wo = (const float*)d_in[4];
    float* out = (float*)d_out;

    char* ws = (char*)d_ws;
    size_t off = 0;
    auto take = [&](size_t bytes) -> char* {
        char* p = ws + off;
        off += (bytes + 255) & ~(size_t)255;
        return p;
    };
    const size_t MT = (size_t)B_ * T_;                 // 16384
    __bf16* xb   = (__bf16*)take(MT * E_ * 2);
    __bf16* WqT  = (__bf16*)take(384 * 384 * 2);
    __bf16* WkT  = (__bf16*)take(384 * 384 * 2);
    __bf16* WvT  = (__bf16*)take(384 * 384 * 2);
    __bf16* WoT  = (__bf16*)take(384 * 384 * 2);
    __bf16* qb   = (__bf16*)take(MT * NQK * 2);        // [B,H,T,HS]
    __bf16* kb   = (__bf16*)take(MT * NQK * 2);        // [B,H,T,HS]
    __bf16* vTb  = (__bf16*)take(MT * NQK * 2);        // [B,H,HS,T]
    __bf16* ctx  = (__bf16*)take(MT * NQK * 2);        // [B*T, 384]

    int nx = (int)(MT * E_);
    k_cvt_x<<<(nx + 255) / 256, 256, 0, stream>>>(x, xb, nx);
    k_prep_w<<<(384 * 384 + 255) / 256, 256, 0, stream>>>(Wq, Wk, Wv, Wo, WqT, WkT, WvT, WoT);
    k_qkv<<<dim3(H_, (int)(MT / 64)), 128, 0, stream>>>(xb, WqT, WkT, WvT, qb, kb, vTb);
    k_attn<<<dim3(B_ * H_, T_ / 64), 128, 0, stream>>>(qb, kb, vTb, ctx);
    k_out<<<dim3(H_, (int)(MT / 64)), 128, 0, stream>>>(ctx, WoT, out);
}
